// GAT_GCN_25383256719663
// MI455X (gfx1250) — matrix operand order repaired and checked by reading, not compiled, not run on hardware
//
#include <hip/hip_runtime.h>
#include <hip/hip_bf16.h>
#include <cmath>

// ---------------- problem constants (from reference) ----------------
static constexpr int HH   = 32;     // heads
static constexpr int CC   = 78;     // channels
static constexpr int DD1  = HH * CC; // 2496
static constexpr int OUTF = 128;
static constexpr int KP1  = 96;     // K=78 padded to multiple of 32 for WMMA

typedef __attribute__((ext_vector_type(16))) __bf16 v16bf;
typedef __attribute__((ext_vector_type(8)))  __bf16 v8bf;
typedef __attribute__((ext_vector_type(8)))  float  v8f;

// ---------------- helpers ----------------
__device__ __forceinline__ unsigned fkey(float f) {
    unsigned u = __float_as_uint(f);
    return (u & 0x80000000u) ? ~u : (u | 0x80000000u);
}
__device__ __forceinline__ float funkey(unsigned k) {
    unsigned u = (k & 0x80000000u) ? (k & 0x7fffffffu) : ~k;
    return __uint_as_float(u);
}
__device__ __forceinline__ float lrelu(float v) { return v > 0.f ? v : 0.2f * v; }

__device__ __forceinline__ void edge_sd(const int* __restrict__ ei, int E, int e,
                                        int& s, int& d) {
    if (e < E) { s = ei[e]; d = ei[E + e]; }   // edge_index[0]=src, [1]=dst
    else       { s = d = e - E; }              // self loop
}

// ---------------- cast / pad kernels ----------------
__global__ void k_cast_pad_x(const float* __restrict__ x, __bf16* __restrict__ xb, int N) {
    int i = blockIdx.x * blockDim.x + threadIdx.x;
    if (i >= N * KP1) return;
    int n = i / KP1, k = i % KP1;
    xb[i] = (__bf16)(k < CC ? x[n * CC + k] : 0.f);
}
__global__ void k_cast_pad_wgat(const float* __restrict__ w, __bf16* __restrict__ wb) {
    int i = blockIdx.x * blockDim.x + threadIdx.x;
    if (i >= KP1 * DD1) return;
    int k = i / DD1, c = i % DD1;
    wb[i] = (__bf16)(k < CC ? w[k * DD1 + c] : 0.f);
}
__global__ void k_cast_wgcn(const float* __restrict__ w, __bf16* __restrict__ wb) {
    int i = blockIdx.x * blockDim.x + threadIdx.x;
    if (i >= DD1 * DD1) return;
    wb[i] = (__bf16)w[i];
}

// ---------------- WMMA GEMM: C[M,N] = A[M,K](bf16) * B[K,N](bf16) ----------------
// One wave computes a 16(M) x 64(N) strip: 4 accumulators, A fragment reused 4x.
// K, N multiples of 32 / 64. EXEC stays all-ones inside active waves.
template <bool OUT_BF16>
__global__ void k_gemm_wmma(const __bf16* __restrict__ A, const __bf16* __restrict__ B,
                            void* __restrict__ Cout, int M, int N, int K) {
    const int lane = threadIdx.x & 31;
    const int wave = (blockIdx.x * blockDim.x + threadIdx.x) >> 5;
    const int nT = N / 64;
    const int mT = (M + 15) >> 4;
    if (wave >= mT * nT) return;                 // whole-wave uniform exit
    const int m0 = (wave / nT) * 16;
    const int n0 = (wave % nT) * 64;
    const int hh  = lane >> 4;                   // half (0/1)
    const int l15 = lane & 15;
    int mrow = m0 + l15; if (mrow >= M) mrow = M - 1;   // clamp loads; stores guarded
    const __bf16* arow = A + (size_t)mrow * K + hh * 8;
    v8f c0 = {}, c1 = {}, c2 = {}, c3 = {};
    for (int k = 0; k < K; k += 32) {
        v8bf lo = *(const v8bf*)(arow + k);       // K = k+8h .. k+8h+7
        v8bf hi = *(const v8bf*)(arow + k + 16);  // K = k+16+8h .. +7
        v16bf a;
#pragma unroll
        for (int i = 0; i < 8; ++i) { a[i] = lo[i]; a[8 + i] = hi[i]; }
        const __bf16* bcol = B + (size_t)k * N + n0 + l15;     // lane -> N, element -> K (table order)
        __builtin_prefetch(arow + k + 64, 0, 1);               // global_prefetch_b8
        v16bf b0, b1, b2, b3;
#pragma unroll
        for (int i = 0; i < 16; ++i) {
            int kk = (i < 8) ? (hh * 8 + i) : (16 + hh * 8 + (i - 8));
            size_t o = (size_t)kk * N;
            b0[i] = bcol[o]; b1[i] = bcol[o + 16]; b2[i] = bcol[o + 32]; b3[i] = bcol[o + 48];
        }
        c0 = __builtin_amdgcn_wmma_f32_16x16x32_bf16(false, a, false, b0, (short)0, c0, false, false);
        c1 = __builtin_amdgcn_wmma_f32_16x16x32_bf16(false, a, false, b1, (short)0, c1, false, false);
        c2 = __builtin_amdgcn_wmma_f32_16x16x32_bf16(false, a, false, b2, (short)0, c2, false, false);
        c3 = __builtin_amdgcn_wmma_f32_16x16x32_bf16(false, a, false, b3, (short)0, c3, false, false);
    }
#pragma unroll
    for (int r = 0; r < 8; ++r) {
        int row = m0 + r + hh * 8;               // D layout: vgpr r -> M = r + 8*half
        if (row >= M) continue;
        size_t base = (size_t)row * N + n0 + l15;
        if (OUT_BF16) {
            __bf16* Cb = (__bf16*)Cout;
            Cb[base]      = (__bf16)c0[r];
            Cb[base + 16] = (__bf16)c1[r];
            Cb[base + 32] = (__bf16)c2[r];
            Cb[base + 48] = (__bf16)c3[r];
        } else {
            float* Cf = (float*)Cout;
            Cf[base]      = c0[r];
            Cf[base + 16] = c1[r];
            Cf[base + 32] = c2[r];
            Cf[base + 48] = c3[r];
        }
    }
}

// ---------------- GAT attention scalars: a_src/a_dst [N,H] ----------------
__global__ void k_att(const __bf16* __restrict__ h, const float* __restrict__ att_s,
                      const float* __restrict__ att_d, float* __restrict__ a_s,
                      float* __restrict__ a_d, int N) {
    int i = blockIdx.x * blockDim.x + threadIdx.x;
    if (i >= N * HH) return;
    int n = i / HH, hd = i % HH;
    const __bf16* hp = h + (size_t)n * DD1 + hd * CC;
    const float* as = att_s + hd * CC;
    const float* ad = att_d + hd * CC;
    float s1 = 0.f, s2 = 0.f;
    for (int c = 0; c < CC; ++c) { float v = (float)hp[c]; s1 += v * as[c]; s2 += v * ad[c]; }
    a_s[i] = s1; a_d[i] = s2;
}

__global__ void k_init_emax(unsigned* __restrict__ emax, int n) {
    int i = blockIdx.x * blockDim.x + threadIdx.x;
    if (i < n) emax[i] = fkey(-3.4e38f);
}

// segment max over dst (+ degree count on head 0)
__global__ void k_edge_max(const int* __restrict__ ei, int E, int Etot,
                           const float* __restrict__ a_s, const float* __restrict__ a_d,
                           unsigned* __restrict__ emax, float* __restrict__ deg) {
    int i = blockIdx.x * blockDim.x + threadIdx.x;
    if (i >= Etot * HH) return;
    int e = i / HH, hd = i % HH;
    int s, d; edge_sd(ei, E, e, s, d);
    float v = lrelu(a_s[s * HH + hd] + a_d[d * HH + hd]);
    atomicMax(&emax[d * HH + hd], fkey(v));
    if (hd == 0) atomicAdd(&deg[d], 1.0f);
}

__global__ void k_edge_denom(const int* __restrict__ ei, int E, int Etot,
                             const float* __restrict__ a_s, const float* __restrict__ a_d,
                             const unsigned* __restrict__ emax, float* __restrict__ denom) {
    int i = blockIdx.x * blockDim.x + threadIdx.x;
    if (i >= Etot * HH) return;
    int e = i / HH, hd = i % HH;
    int s, d; edge_sd(ei, E, e, s, d);
    float v = lrelu(a_s[s * HH + hd] + a_d[d * HH + hd]);
    atomicAdd(&denom[d * HH + hd], __expf(v - funkey(emax[d * HH + hd])));
}

// message scatter: one block per edge; alpha recomputed into LDS; bf16 gather of h
__global__ void k_msg(const int* __restrict__ ei, int E,
                      const float* __restrict__ a_s, const float* __restrict__ a_d,
                      const unsigned* __restrict__ emax, const float* __restrict__ denom,
                      const __bf16* __restrict__ h, float* __restrict__ x1f) {
    __shared__ float alpha[HH];
    int e = blockIdx.x;
    int s, d; edge_sd(ei, E, e, s, d);
    int t = threadIdx.x;
    if (t < HH) {
        float v  = lrelu(a_s[s * HH + t] + a_d[d * HH + t]);
        float ex = __expf(v - funkey(emax[d * HH + t]));
        alpha[t] = ex / denom[d * HH + t];
    }
    __syncthreads();
    const __bf16* hs = h + (size_t)s * DD1;
    float* xd = x1f + (size_t)d * DD1;
    for (int f = t; f < DD1; f += blockDim.x)
        atomicAdd(&xd[f], (float)hs[f] * alpha[f / CC]);
}

// x1 = relu(x1f + b_gat) -> bf16 (reuses the h buffer)
__global__ void k_bias_relu_cast(const float* __restrict__ x1f, const float* __restrict__ b,
                                 __bf16* __restrict__ x1b, int N) {
    int i = blockIdx.x * blockDim.x + threadIdx.x;
    if (i >= N * DD1) return;
    float v = x1f[i] + b[i % DD1];
    x1b[i] = (__bf16)(v > 0.f ? v : 0.f);
}

__global__ void k_dinv(const float* __restrict__ deg, float* __restrict__ dinv, int N) {
    int i = blockIdx.x * blockDim.x + threadIdx.x;
    if (i >= N) return;
    float dg = deg[i];
    dinv[i] = dg > 0.f ? rsqrtf(dg) : 0.f;
}

// GCN aggregation: x2[d] += xw[s] * (dinv[s]*dinv[d]); one block per edge
__global__ void k_gcn_agg(const int* __restrict__ ei, int E,
                          const float* __restrict__ xw, const float* __restrict__ dinv,
                          float* __restrict__ x2) {
    int e = blockIdx.x;
    int s, d; edge_sd(ei, E, e, s, d);
    float norm = dinv[s] * dinv[d];
    const float* xs = xw + (size_t)s * DD1;
    float* xd = x2 + (size_t)d * DD1;
    for (int f = threadIdx.x; f < DD1; f += blockDim.x)
        atomicAdd(&xd[f], xs[f] * norm);
}

// relu(x2 + b_gcn), pooled per graph (sum + count)
__global__ void k_pool(const float* __restrict__ x2, const float* __restrict__ b,
                       const int* __restrict__ batch, float* __restrict__ gsum,
                       float* __restrict__ cnt, int N) {
    int i = blockIdx.x * blockDim.x + threadIdx.x;
    if (i >= N * DD1) return;
    int n = i / DD1, f = i % DD1;
    float v = x2[i] + b[f];
    v = v > 0.f ? v : 0.f;
    int g = batch[n];
    atomicAdd(&gsum[(size_t)g * DD1 + f], v);
    if (f == 0) atomicAdd(&cnt[g], 1.0f);
}

// out = relu([gmean | gsum] @ W_fc + b_fc)
__global__ void k_fc(const float* __restrict__ gsum, const float* __restrict__ cnt,
                     const float* __restrict__ W, const float* __restrict__ b,
                     float* __restrict__ out, int G) {
    int i = blockIdx.x * blockDim.x + threadIdx.x;
    if (i >= G * OUTF) return;
    int g = i / OUTF, o = i % OUTF;
    float invc = 1.0f / fmaxf(cnt[g], 1.0f);
    float acc = b[o];
    const float* gs = gsum + (size_t)g * DD1;
    for (int d = 0; d < DD1; ++d) {
        float sv = gs[d];
        acc += sv * invc * W[d * OUTF + o] + sv * W[(DD1 + d) * OUTF + o];
    }
    out[i] = acc > 0.f ? acc : 0.f;
}

// ---------------- host side ----------------
static inline size_t align256(size_t v) { return (v + 255) & ~(size_t)255; }

extern "C" void kernel_launch(void* const* d_in, const int* in_sizes, int n_in,
                              void* d_out, int out_size, void* d_ws, size_t ws_size,
                              hipStream_t stream) {
    const float* x      = (const float*)d_in[0];
    const int*   ei     = (const int*)d_in[1];
    const int*   batch  = (const int*)d_in[2];
    const float* W_gat  = (const float*)d_in[4];
    const float* att_s  = (const float*)d_in[5];
    const float* att_d  = (const float*)d_in[6];
    const float* b_gat  = (const float*)d_in[7];
    const float* W_gcn  = (const float*)d_in[8];
    const float* b_gcn  = (const float*)d_in[9];
    const float* W_fc   = (const float*)d_in[10];
    const float* b_fc   = (const float*)d_in[11];
    float* out = (float*)d_out;

    const int N    = in_sizes[0] / CC;
    const int E    = in_sizes[1] / 2;
    const int Etot = E + N;
    const int G    = out_size / OUTF;

    // workspace layout
    char* w = (char*)d_ws;
    size_t cur = 0;
    auto take = [&](size_t bytes) { char* p = w + cur; cur += align256(bytes); return p; };
    __bf16*  h_bf   = (__bf16*)take((size_t)N * DD1 * 2);  // h, later reused as x1_bf16
    float*   x1f    = (float*) take((size_t)N * DD1 * 4);  // x1 accum, later reused as x2
    float*   xw     = (float*) take((size_t)N * DD1 * 4);
    __bf16*  x_bf   = (__bf16*)take((size_t)N * KP1 * 2);
    __bf16*  wgat_b = (__bf16*)take((size_t)KP1 * DD1 * 2);
    __bf16*  wgcn_b = (__bf16*)take((size_t)DD1 * DD1 * 2);
    float*   a_s    = (float*) take((size_t)N * HH * 4);
    float*   a_d    = (float*) take((size_t)N * HH * 4);
    unsigned* emax  = (unsigned*)take((size_t)N * HH * 4);
    float*   denom  = (float*) take((size_t)N * HH * 4);
    float*   deg    = (float*) take((size_t)N * 4);
    float*   dinv   = (float*) take((size_t)N * 4);
    float*   gsum   = (float*) take((size_t)G * DD1 * 4);
    float*   cnt    = (float*) take((size_t)G * 4);
    (void)ws_size;

    const int TB = 256;
    auto blks = [&](long long n) { return (unsigned)((n + TB - 1) / TB); };

    // casts / padding
    k_cast_pad_x   <<<blks((long long)N * KP1), TB, 0, stream>>>(x, x_bf, N);
    k_cast_pad_wgat<<<blks((long long)KP1 * DD1), TB, 0, stream>>>(W_gat, wgat_b);
    k_cast_wgcn    <<<blks((long long)DD1 * DD1), TB, 0, stream>>>(W_gcn, wgcn_b);

    // GEMM1: h = x @ W_gat  (bf16 out)
    {
        long long waves = (long long)((N + 15) / 16) * (DD1 / 64);
        unsigned gb = (unsigned)((waves * 32 + 127) / 128);
        k_gemm_wmma<true><<<gb, 128, 0, stream>>>(x_bf, wgat_b, (void*)h_bf, N, DD1, KP1);
    }

    // attention + segment softmax prep
    k_att<<<blks((long long)N * HH), TB, 0, stream>>>(h_bf, att_s, att_d, a_s, a_d, N);
    k_init_emax<<<blks((long long)N * HH), TB, 0, stream>>>(emax, N * HH);
    hipMemsetAsync(denom, 0, (size_t)N * HH * 4, stream);
    hipMemsetAsync(deg,   0, (size_t)N * 4, stream);
    hipMemsetAsync(x1f,   0, (size_t)N * DD1 * 4, stream);

    k_edge_max  <<<blks((long long)Etot * HH), TB, 0, stream>>>(ei, E, Etot, a_s, a_d, emax, deg);
    k_edge_denom<<<blks((long long)Etot * HH), TB, 0, stream>>>(ei, E, Etot, a_s, a_d, emax, denom);
    k_msg<<<Etot, TB, 0, stream>>>(ei, E, a_s, a_d, emax, denom, h_bf, x1f);

    // x1 = relu(x1f + b_gat) -> bf16 (overwrites h buffer, safe: h no longer needed)
    k_bias_relu_cast<<<blks((long long)N * DD1), TB, 0, stream>>>(x1f, b_gat, h_bf, N);
    k_dinv<<<blks(N), TB, 0, stream>>>(deg, dinv, N);

    // GEMM2: xw = x1 @ W_gcn  (f32 out)
    {
        long long waves = (long long)((N + 15) / 16) * (DD1 / 64);
        unsigned gb = (unsigned)((waves * 32 + 127) / 128);
        k_gemm_wmma<false><<<gb, 128, 0, stream>>>(h_bf, wgcn_b, (void*)xw, N, DD1, DD1);
    }

    // GCN aggregation into x2 (reuses x1f buffer)
    hipMemsetAsync(x1f, 0, (size_t)N * DD1 * 4, stream);
    k_gcn_agg<<<Etot, TB, 0, stream>>>(ei, E, xw, dinv, x1f);

    // pooling + fc
    hipMemsetAsync(gsum, 0, (size_t)G * DD1 * 4, stream);
    hipMemsetAsync(cnt,  0, (size_t)G * 4, stream);
    k_pool<<<blks((long long)N * DD1), TB, 0, stream>>>(x1f, b_gcn, batch, gsum, cnt, N);
    k_fc<<<blks((long long)G * OUTF), TB, 0, stream>>>(gsum, cnt, W_fc, b_fc, out, G);
}
